// Encoder_85942295593405
// MI455X (gfx1250) — matrix operand order repaired and checked by reading, not compiled, not run on hardware
//
#include <hip/hip_runtime.h>
#include <hip/hip_bf16.h>
#include <stdint.h>

typedef __attribute__((ext_vector_type(16))) _Float16 v16h;
typedef __attribute__((ext_vector_type(8)))  float    v8f;
typedef __attribute__((ext_vector_type(8)))  uint32_t v8u;

// ---------------------------------------------------------------------------
// helpers
// ---------------------------------------------------------------------------
static __device__ inline uint32_t pack2h(float a, float b) {
    _Float16 ha = (_Float16)a, hb = (_Float16)b;
    uint32_t ua = (uint32_t)__builtin_bit_cast(unsigned short, ha);
    uint32_t ub = (uint32_t)__builtin_bit_cast(unsigned short, hb);
    return ua | (ub << 16);
}

static __device__ inline v8f wmma_f16(v16h a, v16h b, v8f c) {
    return __builtin_amdgcn_wmma_f32_16x16x32_f16(
        /*neg_a=*/false, a, /*neg_b=*/false, b,
        /*c_mod=*/(short)0, c, /*reuse_a=*/false, /*reuse_b=*/false);
}

// A fragment (16x32 f16) from LDS; ISA layout: lane r (0..15) row r,
// VGPR v<4 -> K=2v,2v+1 ; v>=4 -> K=16+2(v-4) ; lanes 16..31 add +8 to K.
static __device__ inline v16h load_a_frag(const _Float16* sm, int rowBase,
                                          int strideHalves, int kOff, int lane) {
    const int r  = lane & 15;
    const int hi = lane >> 4;
    const uint32_t* base = (const uint32_t*)sm;
    const int halfIdx = (rowBase + r) * strideHalves + kOff + hi * 8;
    v8u u;
#pragma unroll
    for (int v = 0; v < 8; ++v) {
        const int k0 = (v < 4) ? (2 * v) : (16 + 2 * (v - 4));
        u[v] = base[(halfIdx + k0) >> 1];
    }
    return __builtin_bit_cast(v16h, u);
}

// B fragment (32x16 f16) pre-packed in workspace as [frag][lane][8 x u32]
static __device__ inline v16h load_b_frag(const uint32_t* __restrict__ bfrags,
                                          int fragIdx, int lane) {
    const v8u* p = (const v8u*)(bfrags + ((size_t)fragIdx * 32 + lane) * 8);
    return __builtin_bit_cast(v16h, *p);
}

// ---------------------------------------------------------------------------
// weight packing: f32 KxN row-major -> f16 B-fragment layout
// grid = KB*NT blocks x 256 threads ; fragIdx = kb*NT + nt
// ---------------------------------------------------------------------------
__global__ __launch_bounds__(256) void pack_b_kernel(
    const float* __restrict__ W, uint32_t* __restrict__ dst,
    int K, int N, int NT) {
    const int f    = blockIdx.x;
    const int kb   = f / NT;
    const int t    = threadIdx.x;
    const int lane = t >> 3;
    const int v    = t & 7;
    const int nt   = f - kb * NT;
    const int col  = nt * 16 + (lane & 15);
    const int k0   = kb * 32 + ((v < 4) ? (2 * v) : (16 + 2 * (v - 4))) + ((lane >> 4) << 3);
    const float a0 = (k0     < K) ? W[(size_t)k0       * N + col] : 0.f;
    const float a1 = (k0 + 1 < K) ? W[(size_t)(k0 + 1) * N + col] : 0.f;
    dst[((size_t)f * 32 + lane) * 8 + v] = pack2h(a0, a1);
}

// ---------------------------------------------------------------------------
// sinusoidal positional encoding table (256 x 16) -> f16
// ---------------------------------------------------------------------------
__global__ __launch_bounds__(256) void pe_kernel(_Float16* __restrict__ pe16) {
    const int idx = blockIdx.x * 256 + threadIdx.x;
    if (idx >= 256 * 16) return;
    const int t = idx >> 4;
    const int c = idx & 15;
    const int i = c >> 1;
    const float div = __expf((float)(2 * i) * (-logf(10000.f) / 16.f));
    const float ang = (float)t * div;
    pe16[idx] = (_Float16)((c & 1) ? __cosf(ang) : __sinf(ang));
}

// ---------------------------------------------------------------------------
// one MLP layer (64 rows) via WMMA; wave handles M-tile mt, N-tiles ntB..ntB+3
// N = 128 (NT=8). Writes f16 result to LDS (stride 136) or global (stride 128).
// ---------------------------------------------------------------------------
static __device__ inline void layer_lds(const _Float16* aLds, int aStride, int KB,
                                        const uint32_t* __restrict__ WF,
                                        const float* __restrict__ bias, bool doRelu,
                                        _Float16* dstLds, int mt, int ntB, int lane) {
    v8f acc[4];
#pragma unroll
    for (int j = 0; j < 4; ++j) {
        const float bv = bias[(ntB + j) * 16 + (lane & 15)];
#pragma unroll
        for (int i = 0; i < 8; ++i) acc[j][i] = bv;
    }
    for (int kb = 0; kb < KB; ++kb) {
        const v16h a = load_a_frag(aLds, mt * 16, aStride, kb * 32, lane);
#pragma unroll
        for (int j = 0; j < 4; ++j) {
            const v16h bf = load_b_frag(WF, kb * 8 + ntB + j, lane);
            acc[j] = wmma_f16(a, bf, acc[j]);
        }
    }
    const int r = lane & 15, hi = lane >> 4;
#pragma unroll
    for (int j = 0; j < 4; ++j) {
        const int col = (ntB + j) * 16 + r;
#pragma unroll
        for (int i = 0; i < 8; ++i) {
            float v = acc[j][i];
            if (doRelu) v = v > 0.f ? v : 0.f;
            dstLds[(mt * 16 + i + hi * 8) * 136 + col] = (_Float16)v;
        }
    }
}

static __device__ inline void layer_glob(const _Float16* aLds, int aStride, int KB,
                                         const uint32_t* __restrict__ WF,
                                         const float* __restrict__ bias,
                                         _Float16* __restrict__ gdst, size_t m0,
                                         int mt, int ntB, int lane) {
    v8f acc[4];
#pragma unroll
    for (int j = 0; j < 4; ++j) {
        const float bv = bias[(ntB + j) * 16 + (lane & 15)];
#pragma unroll
        for (int i = 0; i < 8; ++i) acc[j][i] = bv;
    }
    for (int kb = 0; kb < KB; ++kb) {
        const v16h a = load_a_frag(aLds, mt * 16, aStride, kb * 32, lane);
#pragma unroll
        for (int j = 0; j < 4; ++j) {
            const v16h bf = load_b_frag(WF, kb * 8 + ntB + j, lane);
            acc[j] = wmma_f16(a, bf, acc[j]);
        }
    }
    const int r = lane & 15, hi = lane >> 4;
#pragma unroll
    for (int j = 0; j < 4; ++j) {
        const int col = (ntB + j) * 16 + r;
#pragma unroll
        for (int i = 0; i < 8; ++i) {
            float v = acc[j][i];
            v = v > 0.f ? v : 0.f;   // final MLP layers are relu'd
            gdst[(m0 + (size_t)(mt * 16 + i + hi * 8)) * 128 + col] = (_Float16)v;
        }
    }
}

// ---------------------------------------------------------------------------
// encoder: spikes -> emb -> send/recv MLPs, outputs f16 send16/recv16
// grid = 1024 blocks, 64 rows each (M = B*T*N = 65536)
// ---------------------------------------------------------------------------
__global__ __launch_bounds__(256) void encoder_kernel(
    const float* __restrict__ spikes,
    const uint32_t* __restrict__ WembF, const float* __restrict__ b_emb,
    const uint32_t* __restrict__ Ws1F,  const float* __restrict__ bs1,
    const uint32_t* __restrict__ Ws2F,  const float* __restrict__ bs2,
    const uint32_t* __restrict__ Wr1F,  const float* __restrict__ br1,
    const uint32_t* __restrict__ Wr2F,  const float* __restrict__ br2,
    _Float16* __restrict__ send16, _Float16* __restrict__ recv16) {
    __shared__ _Float16 sA0[64 * 40];    // spikes, K padded 16->32
    __shared__ _Float16 sE [64 * 136];   // emb
    __shared__ _Float16 sH [64 * 136];   // hidden
    const int tid = threadIdx.x, lane = tid & 31, wave = tid >> 5;
    const size_t m0 = (size_t)blockIdx.x * 64;

    {   // stage spikes (64 rows x 16 f32) -> f16, zero-pad K 16..31
        const int r = tid >> 2, c = tid & 3;
        const float4 f = ((const float4*)(spikes + (m0 + r) * 16))[c];
        _Float16* d = sA0 + r * 40 + c * 4;
        d[0] = (_Float16)f.x; d[1] = (_Float16)f.y;
        d[2] = (_Float16)f.z; d[3] = (_Float16)f.w;
        _Float16* z = sA0 + r * 40 + 16 + c * 4;
        z[0] = z[1] = z[2] = z[3] = (_Float16)0.f;
    }
    __syncthreads();

    const int mt = wave & 3, ntB = (wave >> 2) * 4;
    layer_lds(sA0, 40, 1, WembF, b_emb, false, sE, mt, ntB, lane);   // emb
    __syncthreads();
    layer_lds(sE, 136, 4, Ws1F, bs1, true, sH, mt, ntB, lane);       // h_s
    __syncthreads();
    layer_glob(sH, 136, 4, Ws2F, bs2, send16, m0, mt, ntB, lane);    // send
    __syncthreads();
    layer_lds(sE, 136, 4, Wr1F, br1, true, sH, mt, ntB, lane);       // h_r
    __syncthreads();
    layer_glob(sH, 136, 4, Wr2F, br2, recv16, m0, mt, ntB, lane);    // recv
}

// ---------------------------------------------------------------------------
// edge kernel: gather+concat+PE fused with 288x288 WMMA GEMM + 288x5 epilogue
// grid = (16, 255, 2) ; block handles 64 edges of one (b, t_out)
// ---------------------------------------------------------------------------
__global__ __launch_bounds__(256) void edge_kernel(
    const _Float16* __restrict__ send16, const _Float16* __restrict__ recv16,
    const _Float16* __restrict__ pe16,
    const uint32_t* __restrict__ Wc1F, const float* __restrict__ bc1,
    const float* __restrict__ Wc2, const float* __restrict__ bc2,
    const int* __restrict__ send_edges, const int* __restrict__ recv_edges,
    float* __restrict__ out) {
    __shared__ _Float16 smA[64 * 296];       // A tile, reused for f16 hidden
    __shared__ float    sW2[288 * 5 + 8];    // Wc2 staged
    const int tid = threadIdx.x, lane = tid & 31, wave = tid >> 5;
    const int eb = blockIdx.x;     // 0..15
    const int t  = blockIdx.y;     // 0..254  (output time index)
    const int b  = blockIdx.z;     // 0..1
    const int e0 = eb * 64;

    for (int i = tid; i < 288 * 5; i += 256) sW2[i] = Wc2[i];

    // --- gather [send(t+1), pe(t+1), recv(t), pe(t)] -> 64 x 288 f16 tile ---
    const _Float16* sendBase = send16 + ((size_t)(b * 256 + t + 1) * 128) * 128;
    const _Float16* recvBase = recv16 + ((size_t)(b * 256 + t    ) * 128) * 128;
    const uint4* peS = (const uint4*)(pe16 + (t + 1) * 16);
    const uint4* peR = (const uint4*)(pe16 + t * 16);
    for (int idx = tid; idx < 64 * 36; idx += 256) {
        const int rr = idx / 36;
        const int q  = idx - rr * 36;
        uint4 val; int dstHalf;
        if (q < 16) {
            const int se = send_edges[e0 + rr];
            val = ((const uint4*)(sendBase + (size_t)se * 128))[q];
            dstHalf = rr * 296 + q * 8;
        } else if (q < 18) {
            val = peS[q - 16];
            dstHalf = rr * 296 + 128 + (q - 16) * 8;
        } else if (q < 34) {
            const int re = recv_edges[e0 + rr];
            val = ((const uint4*)(recvBase + (size_t)re * 128))[q - 18];
            dstHalf = rr * 296 + 144 + (q - 18) * 8;
        } else {
            val = peR[q - 34];
            dstHalf = rr * 296 + 272 + (q - 34) * 8;
        }
        *((uint4*)(smA + dstHalf)) = val;
    }
    __syncthreads();

    // --- hidden = relu(A @ Wc1 + bc1): 4 Mt x 18 Nt x 9 Kb WMMA ---
    const int mt   = wave & 3;
    const int ntB  = (wave >> 2) * 9;
    v8f acc[9];
#pragma unroll
    for (int j = 0; j < 9; ++j) {
        const float bv = bc1[(ntB + j) * 16 + (lane & 15)];
#pragma unroll
        for (int i = 0; i < 8; ++i) acc[j][i] = bv;
    }
    for (int kb = 0; kb < 9; ++kb) {
        const v16h a = load_a_frag(smA, mt * 16, 296, kb * 32, lane);
#pragma unroll
        for (int j = 0; j < 9; ++j) {
            const v16h bf = load_b_frag(Wc1F, kb * 18 + ntB + j, lane);
            acc[j] = wmma_f16(a, bf, acc[j]);
        }
    }
    __syncthreads();   // everyone done reading smA

    {   // relu -> f16 hidden, reuse smA (stride 296)
        const int r = lane & 15, hi = lane >> 4;
#pragma unroll
        for (int j = 0; j < 9; ++j) {
            const int col = (ntB + j) * 16 + r;
#pragma unroll
            for (int i = 0; i < 8; ++i) {
                float v = acc[j][i];
                smA[(mt * 16 + i + hi * 8) * 296 + col] =
                    (_Float16)(v > 0.f ? v : 0.f);
            }
        }
    }
    __syncthreads();

    // --- out = hidden @ Wc2 + bc2 (288 x 5), 320 dots per block ---
    float* outBase = out + ((size_t)((b * 255 + t) * 1024 + e0)) * 5;
    for (int oi = tid; oi < 64 * 5; oi += 256) {
        const int row = oi / 5;
        const int o   = oi - row * 5;
        float sum = bc2[o];
        const uint4* rowp = (const uint4*)(smA + row * 296);
#pragma unroll 4
        for (int kk = 0; kk < 36; ++kk) {
            const uint4 u = rowp[kk];
            const _Float16* hp = (const _Float16*)&u;
#pragma unroll
            for (int j = 0; j < 8; ++j)
                sum += (float)hp[j] * sW2[(kk * 8 + j) * 5 + o];
        }
        outBase[(size_t)row * 5 + o] = sum;
    }
}

// ---------------------------------------------------------------------------
// launcher
// ---------------------------------------------------------------------------
extern "C" void kernel_launch(void* const* d_in, const int* in_sizes, int n_in,
                              void* d_out, int out_size, void* d_ws, size_t ws_size,
                              hipStream_t stream) {
    const float* spikes = (const float*)d_in[0];
    const float* W_emb  = (const float*)d_in[1];
    const float* b_emb  = (const float*)d_in[2];
    const float* Ws1    = (const float*)d_in[3];
    const float* bs1    = (const float*)d_in[4];
    const float* Ws2    = (const float*)d_in[5];
    const float* bs2    = (const float*)d_in[6];
    const float* Wr1    = (const float*)d_in[7];
    const float* br1    = (const float*)d_in[8];
    const float* Wr2    = (const float*)d_in[9];
    const float* br2    = (const float*)d_in[10];
    const float* Wc1    = (const float*)d_in[11];
    const float* bc1    = (const float*)d_in[12];
    const float* Wc2    = (const float*)d_in[13];
    const float* bc2    = (const float*)d_in[14];
    const int* send_edges = (const int*)d_in[15];
    const int* recv_edges = (const int*)d_in[16];
    float* out = (float*)d_out;

    // workspace carve-up (256B aligned)
    uint8_t* ws = (uint8_t*)d_ws;
    auto carve = [&](size_t bytes) {
        void* p = (void*)ws;
        ws += (bytes + 255) & ~(size_t)255;
        return p;
    };
    uint32_t* WembF = (uint32_t*)carve((size_t)1 * 8  * 32 * 8 * 4);
    uint32_t* Ws1F  = (uint32_t*)carve((size_t)4 * 8  * 32 * 8 * 4);
    uint32_t* Ws2F  = (uint32_t*)carve((size_t)4 * 8  * 32 * 8 * 4);
    uint32_t* Wr1F  = (uint32_t*)carve((size_t)4 * 8  * 32 * 8 * 4);
    uint32_t* Wr2F  = (uint32_t*)carve((size_t)4 * 8  * 32 * 8 * 4);
    uint32_t* Wc1F  = (uint32_t*)carve((size_t)9 * 18 * 32 * 8 * 4);
    _Float16* pe16  = (_Float16*)carve((size_t)256 * 16 * 2);
    _Float16* send16 = (_Float16*)carve((size_t)65536 * 128 * 2);
    _Float16* recv16 = (_Float16*)carve((size_t)65536 * 128 * 2);

    // pack weights into WMMA B-fragment layouts
    pack_b_kernel<<<1 * 8,  256, 0, stream>>>(W_emb, WembF, 16,  128, 8);
    pack_b_kernel<<<4 * 8,  256, 0, stream>>>(Ws1,   Ws1F,  128, 128, 8);
    pack_b_kernel<<<4 * 8,  256, 0, stream>>>(Ws2,   Ws2F,  128, 128, 8);
    pack_b_kernel<<<4 * 8,  256, 0, stream>>>(Wr1,   Wr1F,  128, 128, 8);
    pack_b_kernel<<<4 * 8,  256, 0, stream>>>(Wr2,   Wr2F,  128, 128, 8);
    pack_b_kernel<<<9 * 18, 256, 0, stream>>>(Wc1,   Wc1F,  288, 288, 18);
    pe_kernel<<<16, 256, 0, stream>>>(pe16);

    // encoder: spikes -> send16 / recv16 (f16)
    encoder_kernel<<<1024, 256, 0, stream>>>(
        spikes, WembF, b_emb, Ws1F, bs1, Ws2F, bs2, Wr1F, br1, Wr2F, br2,
        send16, recv16);

    // fused gather + classifier
    edge_kernel<<<dim3(16, 255, 2), 256, 0, stream>>>(
        send16, recv16, pe16, Wc1F, bc1, Wc2, bc2, send_edges, recv_edges, out);
}
